// MemoryReader_26774826123748
// MI455X (gfx1250) — matrix operand order repaired and checked by reading, not compiled, not run on hardware
//
#include <hip/hip_runtime.h>

// MemoryReader attention: O[b] = V[b] * softmax_n( K[b]^T Q[b] / sqrt(64) )
// B=4, CK=64, CV=512, N_mem=9216, N_q=2304.
// Pipeline: (1) fp32->bf16 convert V, (2) convert+transpose K and Q into d_ws,
// (3) main kernel: TDM (tensor_load_to_lds) double-buffered K/V staging with
// descriptor-padded LDS rows, one 16x16 score tile per wave, reference-max
// softmax (chunk-0 column max), bf16 WMMA readout with f32 accumulation.

typedef __attribute__((ext_vector_type(16))) __bf16   v16bf;
typedef __attribute__((ext_vector_type(8)))  float    v8f;
typedef __attribute__((ext_vector_type(4)))  unsigned su4;
typedef __attribute__((ext_vector_type(8)))  unsigned su8;

#define B_      4
#define CK_     64
#define CV_     512
#define NMEM    9216
#define MQ      2304
#define NCHUNK  64
#define NCHUNKS (NMEM / NCHUNK)    // 144
#define KV_STRIDE 66               // TDM pad: 1 DWORD per 32-DWORD (64-elem) row
#define P_STRIDE  68

union BF16V { v16bf v; unsigned u[8]; su4 q[2]; };

__device__ __forceinline__ unsigned short f2bf(float f) {
  unsigned u = __float_as_uint(f);
  u += 0x7FFFu + ((u >> 16) & 1u);           // round-to-nearest-even
  return (unsigned short)(u >> 16);
}
__device__ __forceinline__ unsigned pk2bf(float lo, float hi) {
  unsigned r;
  asm("v_cvt_pk_bf16_f32 %0, %1, %2" : "=v"(r) : "v"(lo), "v"(hi));
  return r;
}
__device__ __forceinline__ float bpermf(float x, int srclane) {
  return __int_as_float(__builtin_amdgcn_ds_bpermute(srclane << 2, __float_as_int(x)));
}

// ---- Tensor Data Mover: 2D tile load, 2-byte elements, LDS row padding ----
// D# per CDNA5 ISA ch.8: group0 {count=1, lds_addr, global_addr, type=2},
// group1 {data_size=2B, pad_enable, pad_interval=32 DWORDs, pad_amount=1 DWORD,
//         tensor_dim0/1, tile_dim0/1, tensor_dim0_stride}. Groups 2/3 = NULL (2D).
__device__ __forceinline__ void tdm_load_2d(unsigned lds_addr, const void* gptr,
                                            unsigned tile_d0, unsigned tile_d1,
                                            unsigned tensor_d0, unsigned tensor_d1,
                                            unsigned stride0) {
  unsigned long long ga = (unsigned long long)(uintptr_t)gptr;
  su4 g0;
  g0[0] = 1u;                                            // count=1 (valid), user mode
  g0[1] = lds_addr;                                      // LDS byte address
  g0[2] = (unsigned)ga;                                  // global addr [31:0]
  g0[3] = (unsigned)((ga >> 32) & 0x01FFFFFFull) | (2u << 30);  // [56:32] | type=2
  su8 g1;
  g1[0] = (1u << 16)        // data_size = 2 bytes
        | (1u << 20)        // pad_enable
        | (4u << 22)        // pad_interval: 32 DWORDs (= one 64-elem row)
        | (0u << 25);       // pad_amount: 1 DWORD  -> LDS row stride 66 ushorts
  g1[1] = (tensor_d0 & 0xFFFFu) << 16;                   // atomic_addr=0 | td0 lo
  g1[2] = (tensor_d0 >> 16) | ((tensor_d1 & 0xFFFFu) << 16);
  g1[3] = (tensor_d1 >> 16) | (tile_d0 << 16);
  g1[4] = (tile_d1 & 0xFFFFu);                           // tile_dim1 | tile_dim2=0
  g1[5] = stride0;                                       // tensor_dim0_stride [31:0]
  g1[6] = 0u;
  g1[7] = 0u;
  asm volatile("tensor_load_to_lds %0, %1" :: "s"(g0), "s"(g1) : "memory");
}

// ---- prep kernel 1: V fp32 -> bf16, same [cv][n] layout ----
__global__ __launch_bounds__(256) void convert_v(const float* __restrict__ in,
                                                 unsigned short* __restrict__ out) {
  size_t base = ((size_t)blockIdx.x * 256 + threadIdx.x) * 16;
  float4 a = *(const float4*)(in + base);
  float4 b = *(const float4*)(in + base + 4);
  float4 c = *(const float4*)(in + base + 8);
  float4 d = *(const float4*)(in + base + 12);
  uint4 lo = make_uint4(pk2bf(a.x, a.y), pk2bf(a.z, a.w), pk2bf(b.x, b.y), pk2bf(b.z, b.w));
  uint4 hi = make_uint4(pk2bf(c.x, c.y), pk2bf(c.z, c.w), pk2bf(d.x, d.y), pk2bf(d.z, d.w));
  *(uint4*)(out + base)     = lo;
  *(uint4*)(out + base + 8) = hi;
}

// ---- prep kernel 2: [64][nfull] fp32 -> [nfull][64] bf16 (scaled), LDS transpose ----
__global__ __launch_bounds__(256) void convert_transpose(const float* __restrict__ in,
                                                         unsigned short* __restrict__ out,
                                                         int nfull, float scale) {
  __shared__ unsigned short tile[32 * KV_STRIDE];
  const int tid = threadIdx.x;
  const int ntiles = nfull / 32;
  const int b  = blockIdx.x / ntiles;
  const int nt = blockIdx.x % ntiles;
  const float* inb = in + (size_t)b * CK_ * nfull + nt * 32;
  unsigned short* outb = out + ((size_t)b * nfull + (size_t)nt * 32) * CK_;
  #pragma unroll
  for (int i = 0; i < 8; ++i) {
    int linear = i * 256 + tid;
    int ck = linear >> 5;
    int n  = linear & 31;
    tile[n * KV_STRIDE + ck] = f2bf(inb[(size_t)ck * nfull + n] * scale);
  }
  __syncthreads();
  int n   = tid >> 3;
  int ckp = (tid & 7) * 8;
  uint4 o;
  o.x = *(const unsigned*)&tile[n * KV_STRIDE + ckp + 0];
  o.y = *(const unsigned*)&tile[n * KV_STRIDE + ckp + 2];
  o.z = *(const unsigned*)&tile[n * KV_STRIDE + ckp + 4];
  o.w = *(const unsigned*)&tile[n * KV_STRIDE + ckp + 6];
  *(uint4*)&outb[n * CK_ + ckp] = o;
}

// ---- main attention kernel ----
__global__ __launch_bounds__(256)
void attn_main(const unsigned short* __restrict__ Kbf,   // [b][n][64]
               const unsigned short* __restrict__ Qbf,   // [b][m][64], pre-scaled
               const unsigned short* __restrict__ Vbf,   // [b][cv][n]
               float* __restrict__ out) {
  __shared__ __align__(128) unsigned short KsL[2][NCHUNK * KV_STRIDE];
  __shared__ __align__(128) unsigned short VsL[2][CV_ * KV_STRIDE];
  __shared__ __align__(16)  unsigned short Ps[32 * P_STRIDE];
  __shared__ float colmaxPart[8][16];
  __shared__ float Lpart[8][16];

  const int tid  = threadIdx.x;
  const int lane = tid & 31;
  const int wave = tid >> 5;
  const int lo16 = lane & 15;
  const int hi   = (lane >> 4) & 1;
  const int mtw  = wave >> 2;     // this wave's score m-subtile (0..1)
  const int ntw  = wave & 3;      // this wave's score n-subtile (0..3)

  const int mtiles = MQ / 32;     // 72
  const int b  = blockIdx.x / mtiles;
  const int m0 = (blockIdx.x % mtiles) * 32;

  const unsigned short* Kb = Kbf + (size_t)b * NMEM * CK_;
  const unsigned short* Qb = Qbf + (size_t)b * MQ * CK_;
  const unsigned short* Vb = Vbf + (size_t)b * CV_ * NMEM;

  // loop-invariant Q B-operands (direct 16B global loads, B layout)
  BF16V qb[2];
  {
    const unsigned short* qrow = Qb + (size_t)(m0 + mtw * 16 + lo16) * CK_;
    #pragma unroll
    for (int h = 0; h < 2; ++h) {
      qb[h].q[0] = *(const su4*)&qrow[h * 32 + hi * 16];
      qb[h].q[1] = *(const su4*)&qrow[h * 32 + hi * 16 + 8];
    }
  }

  v8f acc[4][2];
  #pragma unroll
  for (int ct = 0; ct < 4; ++ct)
    #pragma unroll
    for (int mt = 0; mt < 2; ++mt)
      #pragma unroll
      for (int r = 0; r < 8; ++r) acc[ct][mt][r] = 0.0f;

  float lsum = 0.0f;
  float Mref = 0.0f;

  if (wave == 0) {   // prologue: DMA chunk 0
    tdm_load_2d((unsigned)(uintptr_t)&KsL[0][0], Kb, 64, NCHUNK, 64, NMEM, 64);
    tdm_load_2d((unsigned)(uintptr_t)&VsL[0][0], Vb, NCHUNK, CV_, NMEM, CV_, NMEM);
  }

  for (int c = 0; c < NCHUNKS; ++c) {
    const int buf = c & 1;
    if (wave == 0) __builtin_amdgcn_s_wait_tensorcnt(0);
    __syncthreads();                       // chunk c staged; buffer buf^1 free
    if (wave == 0 && c + 1 < NCHUNKS) {    // overlap next chunk's DMA with compute
      tdm_load_2d((unsigned)(uintptr_t)&KsL[buf ^ 1][0],
                  Kb + (size_t)(c + 1) * NCHUNK * CK_, 64, NCHUNK, 64, NMEM, 64);
      tdm_load_2d((unsigned)(uintptr_t)&VsL[buf ^ 1][0],
                  Vb + (size_t)(c + 1) * NCHUNK, NCHUNK, CV_, NMEM, CV_, NMEM);
    }
    const unsigned short* Ksb = &KsL[buf][0];
    const unsigned short* Vsb = &VsL[buf][0];

    // ---- this wave's 16x16 score tile: rows n=ntw*16.., cols m=mtw*16.. ----
    v8f st;
    #pragma unroll
    for (int r = 0; r < 8; ++r) st[r] = 0.0f;
    #pragma unroll
    for (int h = 0; h < 2; ++h) {
      BF16V a;   // A layout: lanes<16 K={0..15}, lanes>=16 K={16..31}
      #pragma unroll
      for (int v = 0; v < 8; ++v) {
        int ck = h * 32 + hi * 16 + v * 2;
        a.u[v] = *(const unsigned*)&Ksb[(ntw * 16 + lo16) * KV_STRIDE + ck];
      }
      st = __builtin_amdgcn_wmma_f32_16x16x32_bf16(false, a.v, false, qb[h].v,
                                                   (short)0, st, false, false);
    }

    if (c == 0) {   // reference max from first chunk's columns (deterministic)
      float cm = st[0];
      #pragma unroll
      for (int r = 1; r < 8; ++r) cm = fmaxf(cm, st[r]);
      cm = fmaxf(cm, bpermf(cm, lane ^ 16));
      if (lane < 16) colmaxPart[wave][lo16] = cm;
      __syncthreads();
      Mref = fmaxf(fmaxf(colmaxPart[mtw * 4 + 0][lo16], colmaxPart[mtw * 4 + 1][lo16]),
                   fmaxf(colmaxPart[mtw * 4 + 2][lo16], colmaxPart[mtw * 4 + 3][lo16]));
    }

    // ---- p = exp2(s - Mref)  (scores pre-scaled by log2e/8) ----
    float ps[8];
    float lsc = 0.0f;
    #pragma unroll
    for (int r = 0; r < 8; ++r) {
      ps[r] = __builtin_amdgcn_exp2f(st[r] - Mref);
      lsc += ps[r];
    }
    lsum += lsc;
    #pragma unroll
    for (int t = 0; t < 4; ++t)   // rows r,r+1 are adjacent n -> packed dword store
      *(unsigned*)&Ps[(mtw * 16 + lo16) * P_STRIDE + ntw * 16 + 8 * hi + 2 * t] =
          pk2bf(ps[2 * t], ps[2 * t + 1]);

    __syncthreads();               // P ready for all waves

    // ---- readout: this wave's 64 CV rows, both m-subtiles, K=64 in 2 halves ----
    BF16V pbj[2][2];
    #pragma unroll
    for (int mt = 0; mt < 2; ++mt)
      #pragma unroll
      for (int j = 0; j < 2; ++j)
        #pragma unroll
        for (int v = 0; v < 8; ++v)
          pbj[mt][j].u[v] =
              *(const unsigned*)&Ps[(mt * 16 + lo16) * P_STRIDE + j * 32 + hi * 16 + 2 * v];
    #pragma unroll
    for (int ct = 0; ct < 4; ++ct) {
      int row = wave * 64 + ct * 16 + lo16;
      #pragma unroll
      for (int j = 0; j < 2; ++j) {
        BF16V av;
        #pragma unroll
        for (int v = 0; v < 8; ++v) {
          int np = j * 32 + hi * 16 + v * 2;
          av.u[v] = *(const unsigned*)&Vsb[row * KV_STRIDE + np];
        }
        #pragma unroll
        for (int mt = 0; mt < 2; ++mt)
          acc[ct][mt] = __builtin_amdgcn_wmma_f32_16x16x32_bf16(false, av.v, false, pbj[mt][j].v,
                                                                (short)0, acc[ct][mt], false, false);
      }
    }
  }

  // ---- deterministic cross-wave denominator reduce + store ----
  lsum += bpermf(lsum, lane ^ 16);
  if (lane < 16) Lpart[wave][lo16] = lsum;
  __syncthreads();
  float inv[2];
  #pragma unroll
  for (int mt = 0; mt < 2; ++mt) {
    float Lm = (Lpart[mt * 4 + 0][lo16] + Lpart[mt * 4 + 1][lo16]) +
               (Lpart[mt * 4 + 2][lo16] + Lpart[mt * 4 + 3][lo16]);
    inv[mt] = 1.0f / Lm;
  }
  float* outb = out + (size_t)b * CV_ * MQ;
  #pragma unroll
  for (int ct = 0; ct < 4; ++ct)
    #pragma unroll
    for (int mt = 0; mt < 2; ++mt)
      #pragma unroll
      for (int r = 0; r < 8; ++r) {
        int cv = wave * 64 + ct * 16 + r + (hi ? 8 : 0);
        int m  = m0 + mt * 16 + lo16;
        outb[(size_t)cv * MQ + m] = acc[ct][mt][r] * inv[mt];
      }
}

extern "C" void kernel_launch(void* const* d_in, const int* in_sizes, int n_in,
                              void* d_out, int out_size, void* d_ws, size_t ws_size,
                              hipStream_t stream) {
  (void)in_sizes; (void)n_in; (void)out_size; (void)ws_size;
  const float* mk = (const float*)d_in[0];
  const float* qk = (const float*)d_in[1];
  const float* mv = (const float*)d_in[2];
  float* out = (float*)d_out;

  // workspace layout (bf16): V | K^T | Q^T   (~43.6 MB)
  unsigned short* Vbf = (unsigned short*)d_ws;
  unsigned short* Kbf = Vbf + (size_t)B_ * CV_ * NMEM;
  unsigned short* Qbf = Kbf + (size_t)B_ * NMEM * CK_;

  const float qscale = 0.125f * 1.4426950408889634f;   // 1/sqrt(CK) * log2(e)
  hipLaunchKernelGGL(convert_v, dim3((B_ * CV_ * NMEM) / (256 * 16)), dim3(256), 0, stream,
                     mv, Vbf);
  hipLaunchKernelGGL(convert_transpose, dim3(B_ * (NMEM / 32)), dim3(256), 0, stream,
                     mk, Kbf, NMEM, 1.0f);
  hipLaunchKernelGGL(convert_transpose, dim3(B_ * (MQ / 32)), dim3(256), 0, stream,
                     qk, Qbf, MQ, qscale);
  hipLaunchKernelGGL(attn_main, dim3(B_ * (MQ / 32)), dim3(256), 0, stream,
                     Kbf, Qbf, Vbf, out);
}
